// Torch_A16W8SYM_17927193493750
// MI455X (gfx1250) — matrix operand order repaired and checked by reading, not compiled, not run on hardware
//
#include <hip/hip_runtime.h>

typedef __attribute__((ext_vector_type(16))) _Float16 v16h;
typedef __attribute__((ext_vector_type(8)))  _Float16 v8h;
typedef __attribute__((ext_vector_type(4)))  _Float16 v4h;
typedef __attribute__((ext_vector_type(2)))  _Float16 v2h;
typedef __attribute__((ext_vector_type(8)))  float    v8f;
typedef __attribute__((ext_vector_type(4)))  float    f32x4;
typedef __attribute__((ext_vector_type(4)))  int      i32x4;

constexpr int BM = 128, BN = 128, BK = 64;
constexpr int LDK = BK + 8;              // padded f16 row stride (144 B): 16B-aligned, bank-skewed
constexpr int WAVES_M = 2, WAVES_N = 4;  // 8 waves (wave32) per 256-thread block
constexpr int WM = BM / WAVES_M;         // 64 rows per wave
constexpr int WN = BN / WAVES_N;         // 32 cols per wave
constexpr int TM = WM / 16;              // 4 WMMA m-tiles
constexpr int TN = WN / 16;              // 2 WMMA n-tiles

struct h8x2 { v8h lo, hi; };
static __device__ __forceinline__ v16h cat16(v8h lo, v8h hi) {
  h8x2 t{lo, hi};
  return __builtin_bit_cast(v16h, t);
}

static __device__ __forceinline__ v2h pk_f16(float a, float b) {
  return __builtin_bit_cast(v2h, __builtin_amdgcn_cvt_pkrtz(a, b));
}

__global__ __launch_bounds__(256)
void a16w8_wmma_gemm(const float* __restrict__ X, const int* __restrict__ Wq,
                     const float* __restrict__ scales, const float* __restrict__ bias,
                     float* __restrict__ Out, int M, int N, int K) {
  __shared__ _Float16 As[BM * LDK];
  __shared__ _Float16 Bs[BN * LDK];

  const int tid  = threadIdx.x;
  const int lane = tid & 31;
  const int wave = tid >> 5;
  const int waveM = wave >> 2;     // 0..1
  const int waveN = wave & 3;      // 0..3

  const int rowA0 = blockIdx.y * BM;  // M block offset
  const int rowB0 = blockIdx.x * BN;  // N block offset

  const int lsel = lane >> 4;   // half-wave select (0/1)
  const int lrow = lane & 15;

  // Block-invariant epilogue terms: hoist loads + reciprocal divides so they
  // overlap the main loop instead of sitting in the epilogue critical path.
  float rs[TN], bb[TN];
#pragma unroll
  for (int tn = 0; tn < TN; ++tn) {
    const int col = rowB0 + waveN * WN + tn * 16 + lrow;
    rs[tn] = 1.0f / scales[col];
    bb[tn] = bias[col];
  }

  const v8f vzero = {0.f, 0.f, 0.f, 0.f, 0.f, 0.f, 0.f, 0.f};
  v8f acc[TM][TN];
#pragma unroll
  for (int i = 0; i < TM; ++i)
#pragma unroll
    for (int j = 0; j < TN; ++j) acc[i][j] = vzero;

  const int ktiles = K / BK;
  for (int kt = 0; kt < ktiles; ++kt) {
    const int kb = kt * BK;

    // ---- global -> regs (b128 loads), 2048 4-elem chunks over 256 threads ----
    f32x4 ar[8];
    i32x4 br[8];
#pragma unroll
    for (int i = 0; i < 8; ++i) {
      const int c   = tid + i * 256;
      const int row = c >> 4;            // BK/4 = 16 chunks per row
      const int kc  = (c & 15) << 2;
      ar[i] = *(const f32x4*)(X  + (size_t)(rowA0 + row) * K + kb + kc);
      br[i] = *(const i32x4*)(Wq + (size_t)(rowB0 + row) * K + kb + kc);
    }
    if (kt + 1 < ktiles) {               // pull next K-slab toward GL2
      __builtin_prefetch(X  + (size_t)(rowA0 + (tid >> 1)) * K + kb + BK, 0, 2);
      __builtin_prefetch(Wq + (size_t)(rowB0 + (tid >> 1)) * K + kb + BK, 0, 2);
    }

    __syncthreads();                     // previous tile's LDS consumers done
#pragma unroll
    for (int i = 0; i < 8; ++i) {
      const int c   = tid + i * 256;
      const int row = c >> 4;
      const int kc  = (c & 15) << 2;
      // packed f32->f16 (v_cvt_pk_rtz_f16_f32): 2 elems/op
      v2h a01 = pk_f16(ar[i].x, ar[i].y);
      v2h a23 = pk_f16(ar[i].z, ar[i].w);
      v4h ha = { a01.x, a01.y, a23.x, a23.y };
      // int8-range weights: exact via v_cvt_f16_i16
      v4h hb = { (_Float16)(short)br[i].x, (_Float16)(short)br[i].y,
                 (_Float16)(short)br[i].z, (_Float16)(short)br[i].w };
      *(v4h*)(As + row * LDK + kc) = ha;
      *(v4h*)(Bs + row * LDK + kc) = hb;
    }
    __syncthreads();

    // ---- two k=32 WMMA steps per LDS tile ----
#pragma unroll
    for (int ks = 0; ks < 2; ++ks) {
      v16h afr[TM];
#pragma unroll
      for (int tm = 0; tm < TM; ++tm) {
        // A 16x32 f16 layout: half-wave 0 -> K {0..7,16..23}, half-wave 1 -> K {8..15,24..31}
        const int r = waveM * WM + tm * 16 + lrow;
        const _Float16* p = As + r * LDK + ks * 32 + lsel * 8;
        afr[tm] = cat16(*(const v8h*)p, *(const v8h*)(p + 16));
      }
      v16h bfr[TN];
#pragma unroll
      for (int tn = 0; tn < TN; ++tn) {
        // B 32x16 f16 layout: lane n holds column n, half-wave 0 -> K {0..7,16..23}, half-wave 1 -> K {8..15,24..31}
        const int c = waveN * WN + tn * 16 + lrow;
        const _Float16* p = Bs + c * LDK + ks * 32 + lsel * 8;
        bfr[tn] = cat16(*(const v8h*)p, *(const v8h*)(p + 16));
      }
#pragma unroll
      for (int tm = 0; tm < TM; ++tm)
#pragma unroll
        for (int tn = 0; tn < TN; ++tn)
          acc[tm][tn] = __builtin_amdgcn_wmma_f32_16x16x32_f16(
              false, afr[tm], false, bfr[tn], (short)0, acc[tm][tn], false, false);
    }
  }

  // ---- epilogue: out = acc * (1/scale) + bias ----
#pragma unroll
  for (int tn = 0; tn < TN; ++tn) {
    const int col = rowB0 + waveN * WN + tn * 16 + lrow;
#pragma unroll
    for (int tm = 0; tm < TM; ++tm) {
      const int row0 = rowA0 + waveM * WM + tm * 16 + lsel * 8;  // C/D: half-wave 1 = rows +8
#pragma unroll
      for (int r = 0; r < 8; ++r) {
        Out[(size_t)(row0 + r) * N + col] = acc[tm][tn][r] * rs[tn] + bb[tn];
      }
    }
  }
}

extern "C" void kernel_launch(void* const* d_in, const int* in_sizes, int n_in,
                              void* d_out, int out_size, void* d_ws, size_t ws_size,
                              hipStream_t stream) {
  (void)n_in; (void)out_size; (void)d_ws; (void)ws_size;
  const float* x      = (const float*)d_in[0];
  const int*   Wq     = (const int*)d_in[1];
  const float* scales = (const float*)d_in[2];
  const float* bias   = (const float*)d_in[3];
  float* out = (float*)d_out;

  const int N = in_sizes[2];            // 16384 (scales length)
  const int K = in_sizes[1] / N;        // 4096
  const int M = in_sizes[0] / K;        // B*S = 8192

  dim3 grid(N / BN, M / BM);            // 128 x 64 blocks
  a16w8_wmma_gemm<<<grid, dim3(256), 0, stream>>>(x, Wq, scales, bias, out, M, N, K);
}
